// LSTMModel_69741678953132
// MI455X (gfx1250) — matrix operand order repaired and checked by reading, not compiled, not run on hardware
//
#include <hip/hip_runtime.h>
#include <hip/hip_bf16.h>
#include <math.h>

// Problem constants (match reference)
#define Bsz   128
#define Tlen  512
#define INd   128
#define Hd    512
#define G4    2048   // 4*H
#define OUTW  14
#define OUTD  128

typedef _Float16 half8  __attribute__((ext_vector_type(8)));
typedef _Float16 half16 __attribute__((ext_vector_type(16)));
typedef float    float8 __attribute__((ext_vector_type(8)));

// ---------------------------------------------------------------------------
// WMMA fragment loaders, matched to the CDNA5 ISA VGPR layouts (CDNA5 ISA).
//
// A (16x32 f16, M x K): lane L holds row m0+(L&15); halves 0..7 = K[k0+ko ..],
// halves 8..15 = K[k0+16+ko ..] where ko = (L>>4)*8.  -> two b128 loads.
// ---------------------------------------------------------------------------
__device__ __forceinline__ half16 load_a_frag(const _Float16* base, int stride,
                                              int m0, int k0, int lane) {
  const int row = m0 + (lane & 15);
  const int ko  = (lane >> 4) << 3;               // 0 or 8
  const _Float16* p = base + (size_t)row * stride + k0 + ko;
  half8 lo = *(const half8*)(p);
  half8 hi = *(const half8*)(p + 16);
  half16 r;
#pragma unroll
  for (int i = 0; i < 8; ++i) { r[i] = lo[i]; r[i + 8] = hi[i]; }
  return r;
}

// B (32x16 f16, K x N) where B = W^T and W is row-major [N][K]:
// lane L holds column n0+(L&15) = row n0+(L&15) of W; halves 0..7 = K[k0+ko ..],
// halves 8..15 = K[k0+16+ko ..] where ko = (L>>4)*8.  -> two b128 loads.
__device__ __forceinline__ half16 load_b_frag(const _Float16* w, int ldw,
                                              int n0, int k0, int lane) {
  const int row = n0 + (lane & 15);
  const int ko  = (lane >> 4) << 3;               // 0 or 8
  const _Float16* p = w + (size_t)row * ldw + k0 + ko;
  half8 lo = *(const half8*)(p);
  half8 hi = *(const half8*)(p + 16);
  half16 r;
#pragma unroll
  for (int i = 0; i < 8; ++i) { r[i] = lo[i]; r[i + 8] = hi[i]; }
  return r;
}

__device__ __forceinline__ float8 wmma_f16(half16 a, half16 b, float8 c) {
  return __builtin_amdgcn_wmma_f32_16x16x32_f16(false, a, false, b,
                                                (short)0, c, false, false);
}

__device__ __forceinline__ float sigmoid_(float x) {
  return 1.0f / (1.0f + __expf(-x));
}

// ---------------------------------------------------------------------------
// Small utility kernels
// ---------------------------------------------------------------------------
__global__ void f32_to_f16_kernel(const float* __restrict__ in,
                                  _Float16* __restrict__ out, int n) {
  int i = blockIdx.x * blockDim.x + threadIdx.x;
  if (i < n) out[i] = (_Float16)in[i];
}

__global__ void bias_sum_kernel(const float* __restrict__ a,
                                const float* __restrict__ b,
                                float* __restrict__ out, int n) {
  int i = blockIdx.x * blockDim.x + threadIdx.x;
  if (i < n) out[i] = a[i] + b[i];
}

__global__ void zero_state_kernel(float* __restrict__ c,
                                  _Float16* __restrict__ h, int n) {
  int i = blockIdx.x * blockDim.x + threadIdx.x;
  if (i < n) { c[i] = 0.0f; h[i] = (_Float16)0.0f; }
}

// ---------------------------------------------------------------------------
// One LSTM timestep:  gates = [x_t | h_prev] @ [w_ih | w_hh]^T + bias,
// then pointwise LSTM update.  Each wave owns a 16(batch) x 16(hidden) tile
// and accumulates all four gate tiles (i,f,g,o) so the nonlinearity and the
// c/h update are fully lane-local (C/D layout: col = lane&15,
// rows = r + 8*(lane>>4)).
//
// Launch geometry: 256 tiles = 256 waves, packed 2 waves/block (64 threads)
// across 128 blocks so the serialized recurrence spreads over ~128 WGPs
// (minimum per-step tail latency; weights are L2-resident so aggregate L1
// bandwidth scales with WGP count).
// ---------------------------------------------------------------------------
template <int KIN>
__global__ __launch_bounds__(64) void lstm_step_kernel(
    const _Float16* __restrict__ xin, int sx,        // layer input at time t
    const _Float16* __restrict__ w_ih,               // [4H, KIN] f16 row-major
    const _Float16* __restrict__ w_hh,               // [4H, H]   f16 row-major
    const float* __restrict__ bias,                  // [4H] = b_ih + b_hh
    const _Float16* __restrict__ h_prev,             // [B, H] f16
    _Float16* __restrict__ h_next,                   // [B, H] f16
    float* __restrict__ c,                           // [B, H] f32 (in/out)
    _Float16* __restrict__ seq_out, int s_out)       // optional h sink
{
  const int lane = threadIdx.x & 31;
  const int wave = blockIdx.x * (blockDim.x >> 5) + (threadIdx.x >> 5);
  const int m0 = (wave >> 5) << 4;   // batch-row tile   (8 tiles)
  const int n0 = (wave & 31) << 4;   // hidden-col tile  (32 tiles)

  float8 acc[4] = {};

  // Input contribution: K = KIN
  for (int k = 0; k < KIN; k += 32) {
    half16 a = load_a_frag(xin, sx, m0, k, lane);
#pragma unroll
    for (int g = 0; g < 4; ++g) {
      half16 b = load_b_frag(w_ih, KIN, n0 + g * Hd, k, lane);
      acc[g] = wmma_f16(a, b, acc[g]);
    }
  }

  // Recurrent contribution: K = H
  for (int k = 0; k < Hd; k += 32) {
    half16 a = load_a_frag(h_prev, Hd, m0, k, lane);
#pragma unroll
    for (int g = 0; g < 4; ++g) {
      half16 b = load_b_frag(w_hh, Hd, n0 + g * Hd, k, lane);
      acc[g] = wmma_f16(a, b, acc[g]);
    }
  }

  // Pointwise LSTM update (PyTorch gate order i,f,g,o)
  const int n    = lane & 15;
  const int mo   = (lane >> 4) << 3;
  const int hcol = n0 + n;
  const float bI = bias[0 * Hd + hcol];
  const float bF = bias[1 * Hd + hcol];
  const float bG = bias[2 * Hd + hcol];
  const float bO = bias[3 * Hd + hcol];

#pragma unroll
  for (int r = 0; r < 8; ++r) {
    const int brow = m0 + mo + r;
    const size_t idx = (size_t)brow * Hd + hcol;
    const float iv = sigmoid_(acc[0][r] + bI);
    const float fv = sigmoid_(acc[1][r] + bF);
    const float gv = tanhf(acc[2][r] + bG);
    const float ov = sigmoid_(acc[3][r] + bO);
    const float cn = fv * c[idx] + iv * gv;
    c[idx] = cn;
    const float hv = ov * tanhf(cn);
    const _Float16 hh = (_Float16)hv;
    h_next[idx] = hh;
    if (seq_out) seq_out[(size_t)brow * s_out + hcol] = hh;
  }
}

// ---------------------------------------------------------------------------
// FC head: out[R=1792, 128] = h1seq[R, 512] @ w_fc^T + b_fc
// 112 row-tiles x 8 col-tiles = 896 tiles; 1 wave per 16x16 tile,
// 2 waves/block -> 448 blocks.
// ---------------------------------------------------------------------------
__global__ __launch_bounds__(64) void fc_kernel(
    const _Float16* __restrict__ hseq,   // [B*OUTW, H] f16 (contiguous rows)
    const _Float16* __restrict__ wfc,    // [OUTD, H] f16 row-major
    const float* __restrict__ bfc,       // [OUTD]
    float* __restrict__ out)             // [B*OUTW, OUTD] f32
{
  const int lane = threadIdx.x & 31;
  const int wave = blockIdx.x * (blockDim.x >> 5) + (threadIdx.x >> 5);
  const int m0 = (wave >> 3) << 4;   // 112 row tiles
  const int n0 = (wave & 7) << 4;    // 8 col tiles

  float8 acc = {};
  for (int k = 0; k < Hd; k += 32) {
    half16 a = load_a_frag(hseq, Hd, m0, k, lane);
    half16 b = load_b_frag(wfc, Hd, n0, k, lane);
    acc = wmma_f16(a, b, acc);
  }

  const int n  = lane & 15;
  const int mo = (lane >> 4) << 3;
  const float bb = bfc[n0 + n];
#pragma unroll
  for (int r = 0; r < 8; ++r)
    out[(size_t)(m0 + mo + r) * OUTD + n0 + n] = acc[r] + bb;
}

// ---------------------------------------------------------------------------
// Host driver: all launches on `stream`, deterministic, graph-capturable.
// ---------------------------------------------------------------------------
extern "C" void kernel_launch(void* const* d_in, const int* in_sizes, int n_in,
                              void* d_out, int out_size, void* d_ws, size_t ws_size,
                              hipStream_t stream) {
  (void)in_sizes; (void)n_in; (void)out_size; (void)ws_size;

  const float* x    = (const float*)d_in[0];
  const float* wih0 = (const float*)d_in[1];
  const float* whh0 = (const float*)d_in[2];
  const float* bih0 = (const float*)d_in[3];
  const float* bhh0 = (const float*)d_in[4];
  const float* wih1 = (const float*)d_in[5];
  const float* whh1 = (const float*)d_in[6];
  const float* bih1 = (const float*)d_in[7];
  const float* bhh1 = (const float*)d_in[8];
  const float* wfc  = (const float*)d_in[9];
  const float* bfc  = (const float*)d_in[10];
  float* out = (float*)d_out;

  // Workspace carve-up (~93 MB total), 256B aligned chunks.
  char* wsp = (char*)d_ws;
  auto carve = [&](size_t bytes) -> char* {
    char* p = wsp;
    wsp += (bytes + 255) & ~(size_t)255;
    return p;
  };
  _Float16* x_h    = (_Float16*)carve((size_t)Bsz * Tlen * INd * 2);   // 16 MB
  _Float16* wih0_h = (_Float16*)carve((size_t)G4 * INd * 2);
  _Float16* whh0_h = (_Float16*)carve((size_t)G4 * Hd * 2);
  _Float16* wih1_h = (_Float16*)carve((size_t)G4 * Hd * 2);
  _Float16* whh1_h = (_Float16*)carve((size_t)G4 * Hd * 2);
  _Float16* wfc_h  = (_Float16*)carve((size_t)OUTD * Hd * 2);
  float*    bias0  = (float*)carve((size_t)G4 * 4);
  float*    bias1  = (float*)carve((size_t)G4 * 4);
  _Float16* h0seq  = (_Float16*)carve((size_t)Bsz * Tlen * Hd * 2);    // 64 MB
  _Float16* h1seq  = (_Float16*)carve((size_t)Bsz * OUTW * Hd * 2);
  _Float16* hA     = (_Float16*)carve((size_t)Bsz * Hd * 2);
  _Float16* hB     = (_Float16*)carve((size_t)Bsz * Hd * 2);
  float*    cbuf   = (float*)carve((size_t)Bsz * Hd * 4);

  const int TPB = 256;
  auto cvt = [&](const float* src, _Float16* dst, int n) {
    f32_to_f16_kernel<<<(n + TPB - 1) / TPB, TPB, 0, stream>>>(src, dst, n);
  };
  cvt(x,    x_h,    Bsz * Tlen * INd);
  cvt(wih0, wih0_h, G4 * INd);
  cvt(whh0, whh0_h, G4 * Hd);
  cvt(wih1, wih1_h, G4 * Hd);
  cvt(whh1, whh1_h, G4 * Hd);
  cvt(wfc,  wfc_h,  OUTD * Hd);
  bias_sum_kernel<<<(G4 + TPB - 1) / TPB, TPB, 0, stream>>>(bih0, bhh0, bias0, G4);
  bias_sum_kernel<<<(G4 + TPB - 1) / TPB, TPB, 0, stream>>>(bih1, bhh1, bias1, G4);

  // 256 tiles per step, 2 waves (64 threads) per block -> 128 blocks/step.
  const int STEP_TPB    = 64;
  const int stepBlocks  = (Bsz / 16) * (Hd / 16) / (STEP_TPB / 32);   // 128

  // ---- Layer 0: input x_h [B, T, IN], output h0seq [B, T, H] (f16) ----
  zero_state_kernel<<<(Bsz * Hd + TPB - 1) / TPB, TPB, 0, stream>>>(cbuf, hA, Bsz * Hd);
  {
    _Float16* hp = hA; _Float16* hn = hB;
    for (int t = 0; t < Tlen; ++t) {
      lstm_step_kernel<INd><<<stepBlocks, STEP_TPB, 0, stream>>>(
          x_h + (size_t)t * INd, Tlen * INd,
          wih0_h, whh0_h, bias0,
          hp, hn, cbuf,
          h0seq + (size_t)t * Hd, Tlen * Hd);
      _Float16* tmp = hp; hp = hn; hn = tmp;
    }
  }

  // ---- Layer 1: input h0seq [B, T, H], keep first OUTW hidden states ----
  zero_state_kernel<<<(Bsz * Hd + TPB - 1) / TPB, TPB, 0, stream>>>(cbuf, hA, Bsz * Hd);
  {
    _Float16* hp = hA; _Float16* hn = hB;
    for (int t = 0; t < Tlen; ++t) {
      _Float16* sink = (t < OUTW) ? (h1seq + (size_t)t * Hd) : (_Float16*)nullptr;
      lstm_step_kernel<Hd><<<stepBlocks, STEP_TPB, 0, stream>>>(
          h0seq + (size_t)t * Hd, Tlen * Hd,
          wih1_h, whh1_h, bias1,
          hp, hn, cbuf,
          sink, OUTW * Hd);
      _Float16* tmp = hp; hp = hn; hn = tmp;
    }
  }

  // ---- FC head: [B*OUTW, H] @ w_fc^T + b_fc -> [B, OUTW, OUTD] ----
  const int fcBlocks = (Bsz * OUTW / 16) * (OUTD / 16) / 2;   // 448
  fc_kernel<<<fcBlocks, 64, 0, stream>>>(h1seq, wfc_h, bfc, out);
}
